// DMNmodel_82300163326312
// MI455X (gfx1250) — matrix operand order repaired and checked by reading, not compiled, not run on hardware
//
#include <hip/hip_runtime.h>
#include <hip/hip_bf16.h>
#include <stdint.h>

// ---------------------------------------------------------------------------
// DMN model on gfx1250 (MI455X): WMMA f16 GEMMs + persistent recurrent GRU.
// Shapes: B=128 T=512 Tq=32 S=64 G=300 Hd=256 GW=512 V=159 Z=1794
// ---------------------------------------------------------------------------

typedef __attribute__((ext_vector_type(16))) _Float16 v16h;
typedef __attribute__((ext_vector_type(8)))  float    v8f;

union Frag16 { v16h v; uint32_t u[8]; _Float16 h[16]; };

__device__ __forceinline__ float sigmoidf_(float x) { return 1.0f / (1.0f + expf(-x)); }

// A fragment (16x32 f16, M x K) from row-major f16 array (LDS or global).
// ISA 7.12.2: lane m = lane&15; VGPR j holds K = 2j + (lane>=16?8:0) + (j>=4?8:0), pair {k,k+1}.
__device__ __forceinline__ v16h load_a_frag(const _Float16* base, int stride, int k0) {
  const int lane = threadIdx.x & 31;
  const int m = lane & 15;
  const int hk = (lane & 16) ? 8 : 0;
  Frag16 f;
#pragma unroll
  for (int j = 0; j < 8; ++j) {
    int k = k0 + 2 * j + hk + ((j >= 4) ? 8 : 0);
    f.u[j] = *(const uint32_t*)(base + (size_t)m * stride + k);
  }
  return f.v;
}

// B fragment (32x16 f16, K x N) with B[k][n] = W[n][k], W row-major (N x Kpad).
// lane n = n0 + (lane&15); VGPR j holds K = 2j + (lane>=16?8:0) + (j>=4?8:0), pair {k,k+1}.
__device__ __forceinline__ v16h load_b_frag(const _Float16* W, int kpad, int n0, int k0) {
  const int lane = threadIdx.x & 31;
  const int n = n0 + (lane & 15);
  const int hk = (lane & 16) ? 8 : 0;
  Frag16 f;
#pragma unroll
  for (int j = 0; j < 8; ++j) {
    int k = k0 + 2 * j + hk + ((j >= 4) ? 8 : 0);
    f.u[j] = *(const uint32_t*)(W + (size_t)n * kpad + k);
  }
  return f.v;
}

__device__ __forceinline__ v8f wmma_f16(v16h a, v16h b, v8f c) {
  return __builtin_amdgcn_wmma_f32_16x16x32_f16(false, a, false, b, (short)0, c, false, false);
}

// ---------------------------------------------------------------------------
// z-feature source for the attention GEMM (fused construction, never stored).
// z = [C | q | m | C*m | C*q | |C-m| | |C-q| | cWm | cWq], Z=1794, pad to 1824.
// ---------------------------------------------------------------------------
struct ZSrc {
  const float* facts; // (B*S, 256)
  const float* qv;    // (B, 256)
  const float* mt;    // (B, 256)
  const float* cwm;   // (B*S)
  const float* cwq;   // (B*S)
};

__device__ __forceinline__ float zfeat(const ZSrc& z, int row, int k) {
  const int b = row >> 6;
  const float* F = z.facts + ((size_t)row << 8);
  const float* Q = z.qv + ((size_t)b << 8);
  const float* M = z.mt + ((size_t)b << 8);
  if (k < 256)  return F[k];
  if (k < 512)  return Q[k - 256];
  if (k < 768)  return M[k - 512];
  if (k < 1024) return F[k - 768] * M[k - 768];
  if (k < 1280) return F[k - 1024] * Q[k - 1024];
  if (k < 1536) return fabsf(F[k - 1280] - M[k - 1280]);
  if (k < 1792) return fabsf(F[k - 1536] - Q[k - 1536]);
  if (k == 1792) return z.cwm[row];
  if (k == 1793) return z.cwq[row];
  return 0.0f;
}

// ---------------------------------------------------------------------------
// Generic tiled WMMA GEMM: C[M,N] = act(A[M,Kreal] * W[N,Kpad]^T + bias)
// Block: 256 thr (8 waves), tile 32(M) x 256(N); wave = 16(M) x 64(N).
// AMODE 0: A is f32 row-major (stride Kreal, zero-pad K>=Kreal)
// AMODE 1: A rows built from ZSrc.   ACT 0: none, 1: tanh.
// Requires: M % 32 == 0, N % 256 == 0, Kpad % 32 == 0.
// ---------------------------------------------------------------------------
template <int AMODE, int ACT>
__global__ __launch_bounds__(256) void gemm_wmma_kernel(
    const float* __restrict__ A, const _Float16* __restrict__ W,
    const float* __restrict__ bias, float* __restrict__ C,
    int M, int N, int Kreal, int Kpad, ZSrc zs) {
  __shared__ _Float16 atile[32 * 32];
  const int tid = threadIdx.x;
  const int lane = tid & 31, wid = tid >> 5;
  const int wy = wid >> 2, wx = wid & 3;
  const int m_base = blockIdx.y * 32;
  const int n_base = blockIdx.x * 256 + wx * 64;
  v8f acc[4] = {};
  const int nk = Kpad >> 5;
  for (int kk = 0; kk < nk; ++kk) {
    const int k0 = kk << 5;
    for (int i = tid; i < 32 * 32; i += 256) {
      const int mm = i >> 5, kc = i & 31;
      const int gm = m_base + mm, gk = k0 + kc;
      float v;
      if (AMODE == 0) v = (gk < Kreal) ? A[(size_t)gm * Kreal + gk] : 0.0f;
      else            v = zfeat(zs, gm, gk);
      atile[i] = (_Float16)v;
    }
    __syncthreads();
    v16h af = load_a_frag(atile + wy * 16 * 32, 32, 0);
#pragma unroll
    for (int t = 0; t < 4; ++t) {
      v16h bf = load_b_frag(W, Kpad, n_base + t * 16, k0);
      acc[t] = wmma_f16(af, bf, acc[t]);
    }
    __syncthreads();
  }
  const int hr = (lane & 16) ? 8 : 0;
#pragma unroll
  for (int t = 0; t < 4; ++t) {
    const int n = n_base + t * 16 + (lane & 15);
    const float bv = bias ? bias[n] : 0.0f;
#pragma unroll
    for (int r = 0; r < 8; ++r) {
      const int m = m_base + wy * 16 + r + hr;
      float v = acc[t][r] + bv;
      if (ACT == 1) v = tanhf(v);
      C[(size_t)m * N + n] = v;
    }
  }
}

// ---------------------------------------------------------------------------
// Persistent recurrent GRU. 8 workgroups x 16 batch rows; T steps in-kernel.
// Per step: gsum(16x768) = [x|h](16x576) @ Wstack^T  (stacked Wih|Whh, K=576)
//           ghn (16x256) = h @ Whh_n^T               (rows 512..767 of Wstack)
// gates:  r=sig(gs_r+b), z=sig(gs_z+b), hn=ghn+bhh_n,
//         n=tanh(gs_n+b - hn + r*hn),  h' = (1-z)n + z h
// Writes out[b][s][:] whenever sel[b][s] == t (facts gather / q gather).
// ---------------------------------------------------------------------------
__global__ __launch_bounds__(256) void gru_seq_kernel(
    const float* __restrict__ X, int T, int Gin,
    const _Float16* __restrict__ Wstack,  // 768 x 576 f16
    const float* __restrict__ bsum,       // bih+bhh (768)
    const float* __restrict__ bhh,        // bhh (768), uses [512..767]
    const int* __restrict__ sel, int SEL,
    float* __restrict__ out)              // B x SEL x 256
{
  __shared__ _Float16 xh[16 * 576];    // [x(0..319) | h(320..575)]
  __shared__ _Float16 gsum[16 * 768];
  __shared__ _Float16 ghn[16 * 256];
  __shared__ int sel_s[16 * 64];
  const int tid = threadIdx.x, lane = tid & 31, wid = tid >> 5;
  const int b0 = blockIdx.x * 16;

  for (int i = tid; i < 16 * SEL; i += 256) {
    const int m = i / SEL, s = i % SEL;
    sel_s[m * 64 + s] = sel[(b0 + m) * SEL + s];
  }
  for (int i = tid; i < 16 * 256; i += 256) {
    const int m = i >> 8, j = i & 255;
    xh[m * 576 + 320 + j] = (_Float16)0.0f;
  }
  __syncthreads();

  for (int t = 0; t < T; ++t) {
    // stage x_t (zero-pad cols >= Gin)
    for (int i = tid; i < 16 * 320; i += 256) {
      const int m = i / 320, k = i % 320;
      const float v = (k < Gin) ? X[((size_t)(b0 + m) * T + t) * Gin + k] : 0.0f;
      xh[m * 576 + k] = (_Float16)v;
    }
    __syncthreads();

    // main GEMM: wave wid covers n tiles [wid*96, wid*96+96)
    v8f acc[6] = {};
    for (int kk = 0; kk < 18; ++kk) {
      const int k0 = kk * 32;
      v16h af = load_a_frag(xh, 576, k0);
#pragma unroll
      for (int u = 0; u < 6; ++u) {
        v16h bf = load_b_frag(Wstack, 576, wid * 96 + u * 16, k0);
        acc[u] = wmma_f16(af, bf, acc[u]);
      }
    }
    // aux GEMM ghn: h-part only (K in [320,576)), Wstack rows 512+n
    v8f accn[2] = {};
    for (int kk = 0; kk < 8; ++kk) {
      const int k0 = 320 + kk * 32;
      v16h af = load_a_frag(xh, 576, k0);
#pragma unroll
      for (int u = 0; u < 2; ++u) {
        v16h bf = load_b_frag(Wstack, 576, 512 + wid * 32 + u * 16, k0);
        accn[u] = wmma_f16(af, bf, accn[u]);
      }
    }
    // stage results to LDS
    const int hr = (lane & 16) ? 8 : 0;
#pragma unroll
    for (int u = 0; u < 6; ++u) {
      const int n = wid * 96 + u * 16 + (lane & 15);
#pragma unroll
      for (int r = 0; r < 8; ++r) gsum[(r + hr) * 768 + n] = (_Float16)acc[u][r];
    }
#pragma unroll
    for (int u = 0; u < 2; ++u) {
      const int n = wid * 32 + u * 16 + (lane & 15);
#pragma unroll
      for (int r = 0; r < 8; ++r) ghn[(r + hr) * 256 + n] = (_Float16)accn[u][r];
    }
    __syncthreads();

    // gates + h update + gather
    for (int i = tid; i < 16 * 256; i += 256) {
      const int m = i >> 8, j = i & 255;
      const float gr = (float)gsum[m * 768 + j] + bsum[j];
      const float gz = (float)gsum[m * 768 + 256 + j] + bsum[256 + j];
      const float gn = (float)gsum[m * 768 + 512 + j] + bsum[512 + j];
      const float hn = (float)ghn[m * 256 + j] + bhh[512 + j];
      const float h_old = (float)xh[m * 576 + 320 + j];
      const float r = sigmoidf_(gr);
      const float z = sigmoidf_(gz);
      const float nn = tanhf(gn - hn + r * hn);
      const float h_new = (1.0f - z) * nn + z * h_old;
      xh[m * 576 + 320 + j] = (_Float16)h_new;
      for (int s = 0; s < SEL; ++s)
        if (sel_s[m * 64 + s] == t)
          out[((size_t)((b0 + m) * SEL + s)) * 256 + j] = h_new;
    }
    __syncthreads();
  }
}

// ---------------------------------------------------------------------------
// Small VALU kernels
// ---------------------------------------------------------------------------
__device__ __forceinline__ float wave_sum(float v) {
#pragma unroll
  for (int off = 16; off > 0; off >>= 1) v += __shfl_down(v, off, 32);
  return v;
}

// out[r] = sum_k A[r*K+k] * V[(r/rows_per_b)*K + k]
__global__ void dotrows_kernel(float* __restrict__ out, const float* __restrict__ A,
                               const float* __restrict__ V, int K, int rows_per_b, int rows) {
  const int wid = threadIdx.x >> 5, lane = threadIdx.x & 31;
  const int row = blockIdx.x * 8 + wid;
  if (row >= rows) return;
  const float* a = A + (size_t)row * K;
  const float* v = V + (size_t)(row / rows_per_b) * K;
  float s = 0.0f;
  for (int k = lane; k < K; k += 32) s += a[k] * v[k];
  s = wave_sum(s);
  if (lane == 0) out[row] = s;
}

// out[r] = sigmoid(sum_k H1[r*K+k]*w[k] + b[0])
__global__ void att_kernel(float* __restrict__ out, const float* __restrict__ H1,
                           const float* __restrict__ w, const float* __restrict__ b,
                           int K, int rows) {
  const int wid = threadIdx.x >> 5, lane = threadIdx.x & 31;
  const int row = blockIdx.x * 8 + wid;
  if (row >= rows) return;
  const float* a = H1 + (size_t)row * K;
  float s = 0.0f;
  for (int k = lane; k < K; k += 32) s += a[k] * w[k];
  s = wave_sum(s);
  if (lane == 0) out[row] = sigmoidf_(s + b[0]);
}

// per b: soft = softmax(att[b,:64]); Htc[b,h] = sum_s facts[b,s,h]*soft[s]
__global__ __launch_bounds__(256) void softmax_htc_kernel(
    float* __restrict__ htc, const float* __restrict__ att, const float* __restrict__ facts) {
  __shared__ float w[64];
  __shared__ float inv;
  const int b = blockIdx.x, tid = threadIdx.x;
  if (tid < 64) w[tid] = att[b * 64 + tid];
  __syncthreads();
  if (tid == 0) {
    float mx = w[0];
    for (int s = 1; s < 64; ++s) mx = fmaxf(mx, w[s]);
    float sum = 0.0f;
    for (int s = 0; s < 64; ++s) { w[s] = expf(w[s] - mx); sum += w[s]; }
    inv = 1.0f / sum;
  }
  __syncthreads();
  if (tid < 64) w[tid] *= inv;
  __syncthreads();
  float acc = 0.0f;
  for (int s = 0; s < 64; ++s) acc += facts[((size_t)(b * 64 + s)) * 256 + tid] * w[s];
  htc[b * 256 + tid] = acc;
}

// hnew = (1-z)n + z*hold from separate gi(128x768), gh(128x768)
__global__ void gru_gate_kernel(float* __restrict__ hnew, const float* __restrict__ gi,
                                const float* __restrict__ gh, const float* __restrict__ hold,
                                int total) {
  const int i = blockIdx.x * blockDim.x + threadIdx.x;
  if (i >= total) return;
  const int b = i >> 8, j = i & 255;
  const float* a = gi + (size_t)b * 768;
  const float* h = gh + (size_t)b * 768;
  const float r = sigmoidf_(a[j] + h[j]);
  const float z = sigmoidf_(a[256 + j] + h[256 + j]);
  const float n = tanhf(a[512 + j] + r * h[512 + j]);
  hnew[i] = (1.0f - z) * n + z * hold[i];
}

__global__ void logits_kernel(float* __restrict__ out, const float* __restrict__ A,
                              const float* __restrict__ W, int N, int K, int M) {
  const int i = blockIdx.x * blockDim.x + threadIdx.x;
  if (i >= M * N) return;
  const int b = i / N, v = i % N;
  float s = 0.0f;
  for (int k = 0; k < K; ++k) s += A[(size_t)b * K + k] * W[(size_t)v * K + k];
  out[i] = s;
}

__global__ __launch_bounds__(256) void softmax_row_kernel(float* __restrict__ out,
                                                          const float* __restrict__ in, int N) {
  __shared__ float buf[256];
  __shared__ float inv;
  const int b = blockIdx.x, tid = threadIdx.x;
  if (tid < N) buf[tid] = in[b * N + tid];
  __syncthreads();
  if (tid == 0) {
    float mx = buf[0];
    for (int k = 1; k < N; ++k) mx = fmaxf(mx, buf[k]);
    float s = 0.0f;
    for (int k = 0; k < N; ++k) { buf[k] = expf(buf[k] - mx); s += buf[k]; }
    inv = 1.0f / s;
  }
  __syncthreads();
  if (tid < N) out[b * N + tid] = buf[tid] * inv;
}

// ---------------------------------------------------------------------------
// Weight prep kernels
// ---------------------------------------------------------------------------
__global__ void convert_pad_kernel(_Float16* __restrict__ dst, const float* __restrict__ src,
                                   int N, int Kreal, int Kpad) {
  const int i = blockIdx.x * blockDim.x + threadIdx.x;
  if (i >= N * Kpad) return;
  const int n = i / Kpad, k = i % Kpad;
  dst[i] = (_Float16)((k < Kreal) ? src[(size_t)n * Kreal + k] : 0.0f);
}

// dst(768x576) = [Wih(768xGin) | zero pad to 320 | Whh(768x256)]
__global__ void build_stack_kernel(_Float16* __restrict__ dst, const float* __restrict__ wih,
                                   const float* __restrict__ whh, int Gin) {
  const int i = blockIdx.x * blockDim.x + threadIdx.x;
  if (i >= 768 * 576) return;
  const int n = i / 576, c = i % 576;
  float v;
  if (c < Gin) v = wih[(size_t)n * Gin + c];
  else if (c < 320) v = 0.0f;
  else v = whh[(size_t)n * 256 + (c - 320)];
  dst[i] = (_Float16)v;
}

__global__ void bias_sum_kernel(float* dst, const float* a, const float* b, int n) {
  const int i = blockIdx.x * blockDim.x + threadIdx.x;
  if (i < n) dst[i] = a[i] + b[i];
}

__global__ void qsel_kernel(int* qsel, const int* qlen, int B, int Tq) {
  const int i = blockIdx.x * blockDim.x + threadIdx.x;
  if (i < B) {
    int q = qlen[i] - 1;
    q = q < 0 ? 0 : q;
    q = q > Tq - 1 ? Tq - 1 : q;
    qsel[i] = q;
  }
}

// xt(128x416) = [yt(159) | q(256) | 0]
__global__ void concat_xt_kernel(float* __restrict__ xt, const float* __restrict__ yt,
                                 const float* __restrict__ qv) {
  const int i = blockIdx.x * blockDim.x + threadIdx.x;
  if (i >= 128 * 416) return;
  const int b = i / 416, k = i % 416;
  float v;
  if (k < 159) v = yt[b * 159 + k];
  else if (k < 415) v = qv[b * 256 + (k - 159)];
  else v = 0.0f;
  xt[i] = v;
}

// ---------------------------------------------------------------------------
// Host launch
// ---------------------------------------------------------------------------
static inline char* carve(char*& p, size_t bytes) {
  char* r = p;
  p += (bytes + 255) & ~(size_t)255;
  return r;
}

extern "C" void kernel_launch(void* const* d_in, const int* in_sizes, int n_in,
                              void* d_out, int out_size, void* d_ws, size_t ws_size,
                              hipStream_t stream) {
  const float* X     = (const float*)d_in[0];   // (128,512,300)
  const float* Qx    = (const float*)d_in[1];   // (128,32,300)
  const int*   EOS   = (const int*)d_in[2];     // (128,64)
  const int*   qlen  = (const int*)d_in[3];     // (128,)
  const float* Wih_i = (const float*)d_in[6];
  const float* Whh_i = (const float*)d_in[7];
  const float* bih_i = (const float*)d_in[8];
  const float* bhh_i = (const float*)d_in[9];
  const float* Wih_q = (const float*)d_in[10];
  const float* Whh_q = (const float*)d_in[11];
  const float* bih_q = (const float*)d_in[12];
  const float* bhh_q = (const float*)d_in[13];
  const float* zW    = (const float*)d_in[14];
  const float* gW1   = (const float*)d_in[15];
  const float* gb1   = (const float*)d_in[16];
  const float* gW2   = (const float*)d_in[17];
  const float* gb2   = (const float*)d_in[18];
  const float* Wih_m = (const float*)d_in[19];
  const float* Whh_m = (const float*)d_in[20];
  const float* bih_m = (const float*)d_in[21];
  const float* bhh_m = (const float*)d_in[22];
  const float* Wa    = (const float*)d_in[23];
  const float* Wih_a = (const float*)d_in[24];
  const float* Whh_a = (const float*)d_in[25];
  const float* bih_a = (const float*)d_in[26];
  const float* bhh_a = (const float*)d_in[27];

  char* p = (char*)d_ws;
  _Float16* stk_i  = (_Float16*)carve(p, 768 * 576 * 2);
  _Float16* stk_q  = (_Float16*)carve(p, 768 * 576 * 2);
  float*    bsum_i = (float*)carve(p, 768 * 4);
  float*    bsum_q = (float*)carve(p, 768 * 4);
  _Float16* zW16   = (_Float16*)carve(p, 256 * 256 * 2);
  _Float16* gW1h   = (_Float16*)carve(p, 512 * 1824 * 2);
  _Float16* Wmi16  = (_Float16*)carve(p, 768 * 256 * 2);
  _Float16* Wmh16  = (_Float16*)carve(p, 768 * 256 * 2);
  _Float16* Wai16  = (_Float16*)carve(p, 768 * 416 * 2);
  _Float16* Wah16  = (_Float16*)carve(p, 768 * 256 * 2);
  int*      qsel   = (int*)carve(p, 128 * 4);
  float*    facts  = (float*)carve(p, (size_t)128 * 64 * 256 * 4);
  float*    qv     = (float*)carve(p, 128 * 256 * 4);
  float*    cW     = (float*)carve(p, (size_t)8192 * 256 * 4);
  float*    cWq    = (float*)carve(p, 8192 * 4);
  float*    cWm    = (float*)carve(p, 8192 * 4);
  float*    H1     = (float*)carve(p, (size_t)8192 * 512 * 4);
  float*    attb   = (float*)carve(p, 8192 * 4);
  float*    htc    = (float*)carve(p, 128 * 256 * 4);
  float*    mtA    = (float*)carve(p, 128 * 256 * 4);
  float*    mtB    = (float*)carve(p, 128 * 256 * 4);
  float*    gi     = (float*)carve(p, 128 * 768 * 4);
  float*    gh     = (float*)carve(p, 128 * 768 * 4);
  float*    lgt    = (float*)carve(p, 128 * 159 * 4);
  float*    yt     = (float*)carve(p, 128 * 159 * 4);
  float*    xt     = (float*)carve(p, 128 * 416 * 4);

  ZSrc zd = {};

  // --- weight prep -------------------------------------------------------
  build_stack_kernel<<<(768 * 576 + 255) / 256, 256, 0, stream>>>(stk_i, Wih_i, Whh_i, 300);
  build_stack_kernel<<<(768 * 576 + 255) / 256, 256, 0, stream>>>(stk_q, Wih_q, Whh_q, 300);
  bias_sum_kernel<<<3, 256, 0, stream>>>(bsum_i, bih_i, bhh_i, 768);
  bias_sum_kernel<<<3, 256, 0, stream>>>(bsum_q, bih_q, bhh_q, 768);
  convert_pad_kernel<<<(256 * 256 + 255) / 256, 256, 0, stream>>>(zW16, zW, 256, 256, 256);
  convert_pad_kernel<<<(512 * 1824 + 255) / 256, 256, 0, stream>>>(gW1h, gW1, 512, 1794, 1824);
  convert_pad_kernel<<<(768 * 256 + 255) / 256, 256, 0, stream>>>(Wmi16, Wih_m, 768, 256, 256);
  convert_pad_kernel<<<(768 * 256 + 255) / 256, 256, 0, stream>>>(Wmh16, Whh_m, 768, 256, 256);
  convert_pad_kernel<<<(768 * 416 + 255) / 256, 256, 0, stream>>>(Wai16, Wih_a, 768, 415, 416);
  convert_pad_kernel<<<(768 * 256 + 255) / 256, 256, 0, stream>>>(Wah16, Whh_a, 768, 256, 256);
  qsel_kernel<<<1, 128, 0, stream>>>(qsel, qlen, 128, 32);

  // --- recurrent GRUs (persistent, WMMA) ---------------------------------
  gru_seq_kernel<<<8, 256, 0, stream>>>(X, 512, 300, stk_i, bsum_i, bhh_i, EOS, 64, facts);
  gru_seq_kernel<<<8, 256, 0, stream>>>(Qx, 32, 300, stk_q, bsum_q, bhh_q, qsel, 1, qv);

  // --- loop-invariant attention pieces -----------------------------------
  gemm_wmma_kernel<0, 0><<<dim3(1, 256), 256, 0, stream>>>(facts, zW16, nullptr, cW,
                                                           8192, 256, 256, 256, zd);
  dotrows_kernel<<<1024, 256, 0, stream>>>(cWq, cW, qv, 256, 64, 8192);

  // --- episodic memory loop (m_depth = 3) --------------------------------
  const float* mt_cur = qv;  // Mt starts as q
  float* mt_next = mtA;
  for (int d = 0; d < 3; ++d) {
    dotrows_kernel<<<1024, 256, 0, stream>>>(cWm, cW, mt_cur, 256, 64, 8192);
    ZSrc zs = {facts, qv, mt_cur, cWm, cWq};
    gemm_wmma_kernel<1, 1><<<dim3(2, 256), 256, 0, stream>>>(nullptr, gW1h, gb1, H1,
                                                             8192, 512, 1794, 1824, zs);
    att_kernel<<<1024, 256, 0, stream>>>(attb, H1, gW2, gb2, 512, 8192);
    softmax_htc_kernel<<<128, 256, 0, stream>>>(htc, attb, facts);
    gemm_wmma_kernel<0, 0><<<dim3(3, 4), 256, 0, stream>>>(htc, Wmi16, bih_m, gi,
                                                           128, 768, 256, 256, zd);
    gemm_wmma_kernel<0, 0><<<dim3(3, 4), 256, 0, stream>>>(mt_cur, Wmh16, bhh_m, gh,
                                                           128, 768, 256, 256, zd);
    gru_gate_kernel<<<128, 256, 0, stream>>>(mt_next, gi, gh, mt_cur, 128 * 256);
    mt_cur = mt_next;
    mt_next = (mt_next == mtA) ? mtB : mtA;
  }
  const float* atp = mt_cur;  // final Mt

  // --- answer decoder (a_len = 1) ----------------------------------------
  logits_kernel<<<(128 * 159 + 255) / 256, 256, 0, stream>>>(lgt, atp, Wa, 159, 256, 128);
  softmax_row_kernel<<<128, 256, 0, stream>>>(yt, lgt, 159);
  concat_xt_kernel<<<(128 * 416 + 255) / 256, 256, 0, stream>>>(xt, yt, qv);
  gemm_wmma_kernel<0, 0><<<dim3(3, 4), 256, 0, stream>>>(xt, Wai16, bih_a, gi,
                                                         128, 768, 416, 416, zd);
  gemm_wmma_kernel<0, 0><<<dim3(3, 4), 256, 0, stream>>>(atp, Wah16, bhh_a, gh,
                                                         128, 768, 256, 256, zd);
  float* at2 = (mt_cur == mtA) ? mtB : mtA;
  gru_gate_kernel<<<128, 256, 0, stream>>>(at2, gi, gh, atp, 128 * 256);
  logits_kernel<<<(128 * 159 + 255) / 256, 256, 0, stream>>>(lgt, at2, Wa, 159, 256, 128);
  softmax_row_kernel<<<128, 256, 0, stream>>>((float*)d_out, lgt, 159);

  (void)in_sizes; (void)n_in; (void)out_size; (void)ws_size;
}
